// LSTM_61744449848000
// MI455X (gfx1250) — matrix operand order repaired and checked by reading, not compiled, not run on hardware
//
#include <hip/hip_runtime.h>

// ---------------------------------------------------------------------------
// LSTM forward for MI455X (gfx1250, wave32, WMMA).
//   Phase 1: x_proj GEMM with async-to-LDS B staging + bf16 WMMA (f32 accum)
//   Phase 2: 512 sequential fused step kernels (register-pipelined WMMA)
// ---------------------------------------------------------------------------

typedef __bf16 bf16;
typedef __bf16 v16bf __attribute__((ext_vector_type(16)));
typedef __bf16 v8bf  __attribute__((ext_vector_type(8)));
typedef float  v8f   __attribute__((ext_vector_type(8)));

#define B_   64
#define T_   512
#define I_   1024
#define H_   1024
#define G4_  4096

// ---- workspace layout (bytes) ----
static constexpr size_t OFF_XBF   = 0;                               // 64 MB
static constexpr size_t OFF_WX    = OFF_XBF  + (size_t)B_*T_*I_*2;   // 8 MB
static constexpr size_t OFF_WH    = OFF_WX   + (size_t)G4_*I_*2;     // 8 MB
static constexpr size_t OFF_BIAS  = OFF_WH   + (size_t)G4_*H_*2;     // 16 KB
static constexpr size_t OFF_XPROJ = OFF_BIAS + (size_t)G4_*4;        // 512 MB
static constexpr size_t OFF_H0    = OFF_XPROJ+ (size_t)B_*T_*G4_*4;
static constexpr size_t OFF_H1    = OFF_H0   + (size_t)B_*H_*2;
static constexpr size_t OFF_HF    = OFF_H1   + (size_t)B_*H_*2;
static constexpr size_t OFF_CF    = OFF_HF   + (size_t)B_*H_*4;

static __device__ __forceinline__ bf16 f2bf(float f) {
  unsigned u = __builtin_bit_cast(unsigned, f);
  u += 0x7FFFu + ((u >> 16) & 1u);               // round-to-nearest-even
  unsigned short s = (unsigned short)(u >> 16);
  return __builtin_bit_cast(bf16, s);
}

#define CONCAT16(lo, hi) \
  __builtin_shufflevector(lo, hi, 0,1,2,3,4,5,6,7,8,9,10,11,12,13,14,15)

static __device__ __forceinline__ float sigm(float x) {
  return 1.0f / (1.0f + __expf(-x));
}

// async copy 16 bytes global -> LDS (ASYNCcnt-tracked)
static __device__ __forceinline__ void async_ld16(unsigned lds_off, const void* g) {
  asm volatile("global_load_async_to_lds_b128 %0, %1, off"
               :: "v"(lds_off), "v"(g) : "memory");
}

// ---------------------------------------------------------------------------
// prep kernels
// ---------------------------------------------------------------------------
__global__ void k_cvt_bf16(const float* __restrict__ src, bf16* __restrict__ dst, int n) {
  int i = (blockIdx.x * blockDim.x + threadIdx.x) * 4;
  if (i + 3 < n) {
    float4 v = *(const float4*)(src + i);
    dst[i+0] = f2bf(v.x); dst[i+1] = f2bf(v.y);
    dst[i+2] = f2bf(v.z); dst[i+3] = f2bf(v.w);
  }
}

// src: [I_][H_] row-major W[k][j]  ->  dst[j][k]  (N-major, K contiguous)
__global__ void k_cvt_transpose(const float* __restrict__ src, bf16* __restrict__ dst) {
  int j  = blockIdx.x * 32 + (threadIdx.x & 31);
  int k0 = blockIdx.y * 32 + (threadIdx.x >> 5) * 4;
#pragma unroll
  for (int kk = 0; kk < 4; ++kk) {
    int k = k0 + kk;
    dst[(size_t)j * I_ + k] = f2bf(src[(size_t)k * H_ + j]);
  }
}

__global__ void k_bias(const float* __restrict__ bi, const float* __restrict__ bf,
                       const float* __restrict__ bg, const float* __restrict__ bo,
                       float* __restrict__ dst) {
  int i = blockIdx.x * 256 + threadIdx.x;
  if (i < G4_) {
    const float* p = (i < 1024) ? bi : (i < 2048) ? bf : (i < 3072) ? bg : bo;
    dst[i] = p[i & 1023];
  }
}

__global__ void k_zero(unsigned int* __restrict__ p, unsigned n) {
  unsigned i = blockIdx.x * 256u + threadIdx.x;
  if (i < n) p[i] = 0u;
}

// ---------------------------------------------------------------------------
// Phase 1: x_proj GEMM.  Workgroup = 128 rows x 128 cols, 8 waves.
// Wave w computes rows [16w,16w+16) x all 128 cols (8 N-tiles).
// B chunk (128 cols x 32 K) staged cooperatively with
// global_load_async_to_lds_b128, double-buffered on ASYNCcnt.
// All 16 ds_load_b128 for a chunk issue before the 8-WMMA block so the
// DS wait amortizes once per chunk instead of once per WMMA.
// ---------------------------------------------------------------------------
#define BCOL_STRIDE 40   // bf16 elems per LDS column (80B: conflict-free)

__global__ __launch_bounds__(256) void k_xproj(
    const bf16* __restrict__ A,     // [B*T][I]
    const bf16* __restrict__ Wt,    // [4H][I]  N-major
    const float* __restrict__ bias, // [4H]
    float* __restrict__ out)        // [B*T][4H]
{
  __shared__ bf16 sB[2][128][BCOL_STRIDE];   // 20 KB, double buffered

  const int tid  = threadIdx.x;
  const int lane = tid & 31;
  const int w    = tid >> 5;
  const int half = lane >> 4;
  const int l16  = lane & 15;
  const int rowbase = blockIdx.x * 128 + 16 * w;
  const int colbase = blockIdx.y * 128;

  // staging assignment: thread -> (piece = tid&3, cols tid>>2 and tid>>2 + 64)
  const int s_piece = tid & 3;
  const int s_col0  = tid >> 2;

  v8f acc[8] = {};
  const bf16* ap = A + (size_t)(rowbase + l16) * I_ + 8 * half;

  // ---- stage chunk 0 ----
  {
    const bf16* g0 = Wt + (size_t)(colbase + s_col0) * I_ + s_piece * 8;
    const bf16* g1 = Wt + (size_t)(colbase + s_col0 + 64) * I_ + s_piece * 8;
    async_ld16((unsigned)(uintptr_t)&sB[0][s_col0][s_piece * 8], g0);
    async_ld16((unsigned)(uintptr_t)&sB[0][s_col0 + 64][s_piece * 8], g1);
  }

  // preload A chunk 0 into registers
  v8bf a0 = *(const v8bf*)(ap);
  v8bf a1 = *(const v8bf*)(ap + 16);

  for (int k0 = 0; k0 < I_; k0 += 32) {
    const int cur = (k0 >> 5) & 1;
    const int kn  = k0 + 32;

    v16bf a = CONCAT16(a0, a1);          // consume last chunk's A registers

    if (kn < I_) {           // stage next B chunk + pipeline next A loads
      const bf16* g0 = Wt + (size_t)(colbase + s_col0) * I_ + kn + s_piece * 8;
      const bf16* g1 = Wt + (size_t)(colbase + s_col0 + 64) * I_ + kn + s_piece * 8;
      async_ld16((unsigned)(uintptr_t)&sB[cur ^ 1][s_col0][s_piece * 8], g0);
      async_ld16((unsigned)(uintptr_t)&sB[cur ^ 1][s_col0 + 64][s_piece * 8], g1);
      a0 = *(const v8bf*)(ap + kn);
      a1 = *(const v8bf*)(ap + kn + 16);
      asm volatile("s_wait_asynccnt 2" ::: "memory");   // chunk k0 resident
    } else {
      asm volatile("s_wait_asynccnt 0" ::: "memory");
    }
    __syncthreads();                                    // all waves staged

    // issue ALL fragment loads for this chunk, then the WMMA block
    v16bf bfrag[8];
#pragma unroll
    for (int nt = 0; nt < 8; ++nt) {
      const bf16* bp = &sB[cur][16 * nt + l16][8 * half];
      v8bf blo = *(const v8bf*)(bp);
      v8bf bhi = *(const v8bf*)(bp + 16);
      bfrag[nt] = CONCAT16(blo, bhi);
    }
#pragma unroll
    for (int nt = 0; nt < 8; ++nt) {
      acc[nt] = __builtin_amdgcn_wmma_f32_16x16x32_bf16(
          false, a, false, bfrag[nt], (short)0, acc[nt], false, false);
    }
    __syncthreads();        // done reading sB[cur] before it is restaged
  }

  const int srow = rowbase + 8 * half;   // C: lane<16 -> M=r, lane>=16 -> M=8+r
#pragma unroll
  for (int nt = 0; nt < 8; ++nt) {
    int n = colbase + 16 * nt + l16;
    float bv = bias[n];
#pragma unroll
    for (int r = 0; r < 8; ++r) {
      out[(size_t)(srow + r) * G4_ + n] = acc[nt][r] + bv;
    }
  }
}

// ---------------------------------------------------------------------------
// Phase 2: fused recurrent step.  64 workgroups; wg owns hidden units
// [16*wg, 16*wg+16) across ALL FOUR gates.  Register-pipelined WMMA loop.
// ---------------------------------------------------------------------------
__global__ __launch_bounds__(256) void k_step(
    const bf16* __restrict__ h_in,   // [B][H] bf16
    bf16* __restrict__ h_out,        // [B][H] bf16
    const bf16* __restrict__ Wt,     // [4H][H] N-major bf16
    const float* __restrict__ xproj, // [B*T][4H]
    float* __restrict__ cbuf,        // [B][H]
    float* __restrict__ hbuf,        // [B][H]
    int t)
{
  __shared__ float lg[64][68];

  const int tid  = threadIdx.x;
  const int lane = tid & 31;
  const int w    = tid >> 5;
  const int half = lane >> 4;
  const int l16  = lane & 15;
  const int mtile = w & 3;
  const int pair  = w >> 2;
  const int ubase = blockIdx.x * 16;

  v8f acc[2] = {};
  const bf16* ap  = h_in + (size_t)(16 * mtile + l16) * H_ + 16 * half;
  const bf16* bp0 = Wt + (size_t)((2 * pair + 0) * H_ + ubase + l16) * H_ + 16 * half;
  const bf16* bp1 = Wt + (size_t)((2 * pair + 1) * H_ + ubase + l16) * H_ + 16 * half;

  // preload chunk 0
  v8bf a0 = *(const v8bf*)(ap);
  v8bf a1 = *(const v8bf*)(ap + 8);
  v8bf c0 = *(const v8bf*)(bp0);
  v8bf c1 = *(const v8bf*)(bp0 + 8);
  v8bf d0 = *(const v8bf*)(bp1);
  v8bf d1 = *(const v8bf*)(bp1 + 8);

  for (int k0 = 0; k0 < H_; k0 += 32) {
    v16bf a  = CONCAT16(a0, a1);
    v16bf b0 = CONCAT16(c0, c1);
    v16bf b1 = CONCAT16(d0, d1);
    const int kn = k0 + 32;
    if (kn < H_) {                       // pipeline next chunk's loads
      a0 = *(const v8bf*)(ap + kn);
      a1 = *(const v8bf*)(ap + kn + 8);
      c0 = *(const v8bf*)(bp0 + kn);
      c1 = *(const v8bf*)(bp0 + kn + 8);
      d0 = *(const v8bf*)(bp1 + kn);
      d1 = *(const v8bf*)(bp1 + kn + 8);
      __builtin_prefetch(bp0 + kn + 128, 0, 1);
      __builtin_prefetch(bp1 + kn + 128, 0, 1);
    }
    acc[0] = __builtin_amdgcn_wmma_f32_16x16x32_bf16(
        false, a, false, b0, (short)0, acc[0], false, false);
    acc[1] = __builtin_amdgcn_wmma_f32_16x16x32_bf16(
        false, a, false, b1, (short)0, acc[1], false, false);
  }

  // gates = acc + x_proj[:, t, :]  -> LDS
  const int srow = 16 * mtile + 8 * half;
#pragma unroll
  for (int gi = 0; gi < 2; ++gi) {
    int g = 2 * pair + gi;
    int n = g * 1024 + ubase + l16;
#pragma unroll
    for (int r = 0; r < 8; ++r) {
      float v = acc[gi][r] + xproj[((size_t)(srow + r) * T_ + t) * G4_ + n];
      lg[srow + r][g * 16 + l16] = v;
    }
  }
  __syncthreads();

  // elementwise update: 64 rows x 16 units
  for (int e = tid; e < 64 * 16; e += 256) {
    int m = e >> 4, j = e & 15;
    int u = ubase + j;
    float iv = sigm(lg[m][ 0 + j]);
    float fv = sigm(lg[m][16 + j]);
    float gv = tanhf(lg[m][32 + j]);
    float ov = sigm(lg[m][48 + j]);
    size_t idx = (size_t)m * H_ + u;
    float c = cbuf[idx];
    float cn = fv * c + iv * gv;
    float hn = ov * tanhf(cn);
    cbuf[idx]  = cn;
    hbuf[idx]  = hn;
    h_out[idx] = f2bf(hn);
  }
}

__global__ void k_out(const float* __restrict__ hf, const float* __restrict__ cf,
                      float* __restrict__ out) {
  int i = blockIdx.x * 256 + threadIdx.x;
  if (i < B_ * H_) {
    out[i]           = hf[i];
    out[B_ * H_ + i] = cf[i];
  }
}

// ---------------------------------------------------------------------------
extern "C" void kernel_launch(void* const* d_in, const int* in_sizes, int n_in,
                              void* d_out, int out_size, void* d_ws, size_t ws_size,
                              hipStream_t stream) {
  const float* x   = (const float*)d_in[0];
  const float* Wii = (const float*)d_in[1];
  const float* Whi = (const float*)d_in[2];
  const float* bhi = (const float*)d_in[3];
  const float* Wif = (const float*)d_in[4];
  const float* Whf = (const float*)d_in[5];
  const float* bhf = (const float*)d_in[6];
  const float* Wig = (const float*)d_in[7];
  const float* Whg = (const float*)d_in[8];
  const float* bhg = (const float*)d_in[9];
  const float* Wio = (const float*)d_in[10];
  const float* Who = (const float*)d_in[11];
  const float* bho = (const float*)d_in[12];

  char* ws = (char*)d_ws;
  bf16*  xbf   = (bf16*)(ws + OFF_XBF);
  bf16*  wxt   = (bf16*)(ws + OFF_WX);
  bf16*  wht   = (bf16*)(ws + OFF_WH);
  float* bias  = (float*)(ws + OFF_BIAS);
  float* xproj = (float*)(ws + OFF_XPROJ);
  bf16*  h0    = (bf16*)(ws + OFF_H0);
  bf16*  h1    = (bf16*)(ws + OFF_H1);
  float* hbuf  = (float*)(ws + OFF_HF);
  float* cbuf  = (float*)(ws + OFF_CF);

  // ---- prep ----
  k_cvt_bf16<<<(B_ * T_ * I_) / 1024, 256, 0, stream>>>(x, xbf, B_ * T_ * I_);

  const float* wx_src[4] = {Wii, Wif, Wig, Wio};
  const float* wh_src[4] = {Whi, Whf, Whg, Who};
  for (int g = 0; g < 4; ++g) {
    k_cvt_transpose<<<dim3(H_ / 32, I_ / 32), 256, 0, stream>>>(
        wx_src[g], wxt + (size_t)g * H_ * I_);
    k_cvt_transpose<<<dim3(H_ / 32, I_ / 32), 256, 0, stream>>>(
        wh_src[g], wht + (size_t)g * H_ * I_);
  }
  k_bias<<<16, 256, 0, stream>>>(bhi, bhf, bhg, bho, bias);

  k_zero<<<(B_ * H_ * 2 / 4 + 255) / 256, 256, 0, stream>>>(
      (unsigned int*)h0, B_ * H_ * 2 / 4);
  k_zero<<<(B_ * H_ + 255) / 256, 256, 0, stream>>>(
      (unsigned int*)cbuf, B_ * H_);

  // ---- phase 1 ----
  k_xproj<<<dim3((B_ * T_) / 128, G4_ / 128), 256, 0, stream>>>(xbf, wxt, bias, xproj);

  // ---- phase 2 ----
  bf16* hb[2] = {h0, h1};
  for (int t = 0; t < T_; ++t) {
    k_step<<<H_ / 16, 256, 0, stream>>>(hb[t & 1], hb[(t + 1) & 1], wht,
                                        xproj, cbuf, hbuf, t);
  }

  k_out<<<(B_ * H_ + 255) / 256, 256, 0, stream>>>(hbuf, cbuf, (float*)d_out);
}
